// TalkingHeadAttention_15728170238612
// MI455X (gfx1250) — matrix operand order repaired and checked by reading, not compiled, not run on hardware
//
#include <hip/hip_runtime.h>

// ---------------------------------------------------------------------------
// Talking-head attention, fused for MI455X (gfx1250, wave32, WMMA bf16,
// async global->LDS double buffering for the K/V stream).
// B=4, N=1024, C=512, H=32, hd=16.
// ---------------------------------------------------------------------------

typedef unsigned int   u32;
typedef unsigned short u16;

typedef __attribute__((ext_vector_type(16))) __bf16 v16bf;
typedef __attribute__((ext_vector_type(8)))  float  v8f;

union BFrag { v16bf v; u32 u[8]; u16 s[16]; };
union FFrag { v8f   v; float f[8]; };

#define BB 4
#define NN 1024
#define CC 512
#define HH 32
#define DD 16

// wait until at most `n` async (global<->LDS) ops remain outstanding
#define ASYNC_WAIT(n) asm volatile("s_wait_asynccnt " #n ::: "memory")

__device__ __forceinline__ u16 f2bf(float x) {
  union { float f; u32 u; } a; a.f = x;
  u32 r = (a.u + 0x7FFFu + ((a.u >> 16) & 1u)) >> 16;  // RNE
  return (u16)r;
}

__device__ __forceinline__ v8f zero8() {
  v8f z = {0.f, 0.f, 0.f, 0.f, 0.f, 0.f, 0.f, 0.f};
  return z;
}

__device__ __forceinline__ v8f wmma_bf16(const BFrag& a, const BFrag& b, v8f c) {
  return __builtin_amdgcn_wmma_f32_16x16x32_bf16(false, a.v, false, b.v,
                                                 (short)0, c, false, false);
}

// one async 512B tile-slice copy: 32 lanes x b128 (GVS addressing:
// SGPR64 base + per-lane 32-bit byte offset), tracked by ASYNCcnt
__device__ __forceinline__ void async_copy_b128(u32 lds_off, u32 glb_off,
                                                unsigned long long base) {
  asm volatile("global_load_async_to_lds_b128 %0, %1, %2"
               :: "v"(lds_off), "v"(glb_off), "s"(base)
               : "memory");
}

// ---------------------------------------------------------------------------
// Kernel 1: f32 -> bf16 conversion of x and all weight matrices.
// ---------------------------------------------------------------------------
__global__ __launch_bounds__(256) void convert_kernel(
    const float* __restrict__ x, const float* __restrict__ wqkv,
    const float* __restrict__ wproj, const float* __restrict__ wpre,
    const float* __restrict__ wpost,
    u16* __restrict__ xb, u16* __restrict__ wqkvb, u16* __restrict__ wprojb,
    u16* __restrict__ wpreb, u16* __restrict__ wpostb) {
  const int i0 = blockIdx.x * blockDim.x + threadIdx.x;
  const int stride = gridDim.x * blockDim.x;
  for (int i = i0; i < BB * NN * CC; i += stride) xb[i] = f2bf(x[i]);
  for (int i = i0; i < 3 * CC * CC; i += stride) wqkvb[i] = f2bf(wqkv[i]);
  for (int i = i0; i < CC * CC; i += stride) wprojb[i] = f2bf(wproj[i]);
  if (i0 < HH * HH) {
    wpreb[i0]  = f2bf(wpre[i0]);
    wpostb[i0] = f2bf(wpost[i0]);
  }
}

// ---------------------------------------------------------------------------
// Kernel 2: QKV projection.  qkv = x @ Wqkv^T + bqkv, then scatter into
//   q[b][h][n][d] (pre-scaled by hd^-0.5 = 0.25), k[b][h][n][d], vT[b][h][d][n]
// ---------------------------------------------------------------------------
__global__ __launch_bounds__(256) void qkv_gemm_kernel(
    const u16* __restrict__ xb, const u16* __restrict__ wqkvb,
    const float* __restrict__ bqkv,
    u16* __restrict__ qb, u16* __restrict__ kb, u16* __restrict__ vt) {
  const int w = threadIdx.x >> 5;
  const int lane = threadIdx.x & 31;
  const int lh = lane >> 4, lc = lane & 15;
  const int m0 = blockIdx.x << 4;                 // row tile
  const int nb = blockIdx.y * 512 + w * 64;       // wave's 64-wide N strip

  FFrag acc[4];
#pragma unroll
  for (int j = 0; j < 4; ++j)
#pragma unroll
    for (int r = 0; r < 8; ++r) acc[j].f[r] = 0.f;

  for (int k0 = 0; k0 < CC; k0 += 32) {
    BFrag af;  // A frag: row = lc, K chunks [k0+lh*8, +8) and [k0+16+lh*8, +8)
    {
      const uint4 t0 = *(const uint4*)(xb + (m0 + lc) * CC + k0 + lh * 8);
      const uint4 t1 = *(const uint4*)(xb + (m0 + lc) * CC + k0 + 16 + lh * 8);
      af.u[0] = t0.x; af.u[1] = t0.y; af.u[2] = t0.z; af.u[3] = t0.w;
      af.u[4] = t1.x; af.u[5] = t1.y; af.u[6] = t1.z; af.u[7] = t1.w;
    }
#pragma unroll
    for (int j = 0; j < 4; ++j) {
      const int n = nb + j * 16 + lc;  // B frag: col = lc, K chunks [k0+lh*8, +8) and [k0+16+lh*8, +8)
      const u16* pb = wqkvb + n * CC + k0 + lh * 8;
      const uint4 b0 = *(const uint4*)pb, b1 = *(const uint4*)(pb + 16);
      BFrag bf;
      bf.u[0] = b0.x; bf.u[1] = b0.y; bf.u[2] = b0.z; bf.u[3] = b0.w;
      bf.u[4] = b1.x; bf.u[5] = b1.y; bf.u[6] = b1.z; bf.u[7] = b1.w;
      acc[j].v = wmma_bf16(af, bf, acc[j].v);
    }
  }

#pragma unroll
  for (int j = 0; j < 4; ++j) {
    const int n = nb + j * 16 + lc;     // output feature index in [0, 1536)
    const int ty = n >> 9;              // 0=q, 1=k, 2=v
    const int cc = n & 511;
    const int h = cc >> 4, d = cc & 15;
    const float bias = bqkv[n];
#pragma unroll
    for (int r = 0; r < 8; ++r) {
      const int row = m0 + r + 8 * lh;          // global token row
      const int bi = row >> 10, ni = row & 1023;
      const float val = acc[j].f[r] + bias;
      if (ty == 0)
        qb[((bi * HH + h) * NN + ni) * DD + d] = f2bf(val * 0.25f);
      else if (ty == 1)
        kb[((bi * HH + h) * NN + ni) * DD + d] = f2bf(val);
      else
        vt[((bi * HH + h) * DD + d) * NN + ni] = f2bf(val);
    }
  }
}

// ---------------------------------------------------------------------------
// Kernel 3: fused talking-head attention.
// One block = (b, 16-query tile).  8 waves; wave w owns heads 4w..4w+3 for
// QK^T and for A~@V, and n-rows {2w,2w+1} for the mix/softmax stages.
// K/V tiles stream HBM -> LDS via global_load_async_to_lds_b128 with
// per-wave double buffering (ASYNCcnt in-order completion => no barriers).
// Pass 1: scores -> pre-mix -> online (max,sum) per mixed row, in registers.
// Pass 2: recompute scores/pre-mix, normalize, post-mix, accumulate vs V.
// ---------------------------------------------------------------------------
__global__ __launch_bounds__(256) void attn_kernel(
    const u16* __restrict__ qb, const u16* __restrict__ kb,
    const u16* __restrict__ vt, const u16* __restrict__ wpreb,
    const u16* __restrict__ wpostb, const float* __restrict__ bpre,
    const float* __restrict__ bpost, u16* __restrict__ xab) {
  __shared__ __attribute__((aligned(16))) u16 Sh[256 * 34];       // [pos][head]
  __shared__ __attribute__((aligned(16))) u16 At[HH * 16 * 16];   // [g'][n][m]
  __shared__ __attribute__((aligned(16))) u16 Kt[2][HH * 256];    // [h][m][d]
  __shared__ __attribute__((aligned(16))) u16 Vt2[2][HH * 256];   // [h][d][m]

  const int w = threadIdx.x >> 5;
  const int lane = threadIdx.x & 31;
  const int lh = lane >> 4, lc = lane & 15;
  const int b = blockIdx.x >> 6;
  const int n0 = (blockIdx.x & 63) << 4;

  const unsigned long long kb64 = (unsigned long long)kb;
  const unsigned long long vt64 = (unsigned long long)vt;

  // async issue of this wave's 4 head tiles (512B each -> 1 b128 op / head)
  auto issue_k = [&](int mt, int buf) {
#pragma unroll
    for (int i = 0; i < 4; ++i) {
      const int h = w * 4 + i;
      const u32 goff =
          (u32)((((b * HH + h) * NN + mt * 16) * DD + lane * 8) * 2);
      const u32 loff = (u32)(unsigned long long)&Kt[buf][h * 256 + lane * 8];
      async_copy_b128(loff, goff, kb64);
    }
  };
  auto issue_v = [&](int mt, int buf) {
#pragma unroll
    for (int i = 0; i < 4; ++i) {
      const int gp = w * 4 + i;
      const u32 goff = (u32)(
          (((b * HH + gp) * DD + (lane >> 1)) * NN + mt * 16 + (lane & 1) * 8) *
          2);
      const u32 loff = (u32)(unsigned long long)
          &Vt2[buf][gp * 256 + (lane >> 1) * 16 + (lane & 1) * 8];
      async_copy_b128(loff, goff, vt64);
    }
  };

  // --- persistent register fragments -------------------------------------
  BFrag qf[4];  // q A-frags for heads 4w..4w+3 (K padded 16->32)
#pragma unroll
  for (int i = 0; i < 4; ++i) {
    const int h = w * 4 + i;
    const uint4 t =
        *(const uint4*)(qb + ((b * HH + h) * NN + n0 + lc) * DD + lh * 8);
    qf[i].u[0] = t.x; qf[i].u[1] = t.y; qf[i].u[2] = t.z; qf[i].u[3] = t.w;
    qf[i].u[4] = 0; qf[i].u[5] = 0; qf[i].u[6] = 0; qf[i].u[7] = 0;
  }
  BFrag pref[2], postf[2];  // Wpre/Wpost A-frags, full K=32
#pragma unroll
  for (int gt = 0; gt < 2; ++gt) {
    const int g = gt * 16 + lc;
    {
      const uint4 t0 = *(const uint4*)(wpreb + g * HH + lh * 16);
      const uint4 t1 = *(const uint4*)(wpreb + g * HH + lh * 16 + 8);
      pref[gt].u[0] = t0.x; pref[gt].u[1] = t0.y; pref[gt].u[2] = t0.z;
      pref[gt].u[3] = t0.w; pref[gt].u[4] = t1.x; pref[gt].u[5] = t1.y;
      pref[gt].u[6] = t1.z; pref[gt].u[7] = t1.w;
    }
    {
      const uint4 t0 = *(const uint4*)(wpostb + g * HH + lh * 16);
      const uint4 t1 = *(const uint4*)(wpostb + g * HH + lh * 16 + 8);
      postf[gt].u[0] = t0.x; postf[gt].u[1] = t0.y; postf[gt].u[2] = t0.z;
      postf[gt].u[3] = t0.w; postf[gt].u[4] = t1.x; postf[gt].u[5] = t1.y;
      postf[gt].u[6] = t1.z; postf[gt].u[7] = t1.w;
    }
  }
  float bpre_r[2][8], bpost_r[2][8];
#pragma unroll
  for (int gt = 0; gt < 2; ++gt)
#pragma unroll
    for (int r = 0; r < 8; ++r) {
      bpre_r[gt][r]  = bpre[gt * 16 + 8 * lh + r];
      bpost_r[gt][r] = bpost[gt * 16 + 8 * lh + r];
    }

  float runM[2][2][8], runL[2][2][8];  // [tI][gt][r] : row (g, n = n0+2w+tI)
#pragma unroll
  for (int tI = 0; tI < 2; ++tI)
#pragma unroll
    for (int gt = 0; gt < 2; ++gt)
#pragma unroll
      for (int r = 0; r < 8; ++r) { runM[tI][gt][r] = -1e30f; runL[tI][gt][r] = 0.f; }

  const u32* Sh32 = (const u32*)Sh;

  // scores for this wave's 4 heads (K tile from LDS) -> Sh[pos=n*16+m][h]
  auto stage_scores = [&](int buf) {
#pragma unroll
    for (int i = 0; i < 4; ++i) {
      const int h = w * 4 + i;
      BFrag kf;
      {  // B frag: elems 0-7 -> d = lh*8+e, elems 8-15 (K=16..31) zero pad
        const uint4 t0 = *(const uint4*)&Kt[buf][h * 256 + lc * 16 + lh * 8];
        kf.u[0] = t0.x; kf.u[1] = t0.y; kf.u[2] = t0.z; kf.u[3] = t0.w;
        kf.u[4] = 0; kf.u[5] = 0; kf.u[6] = 0; kf.u[7] = 0;
      }
      FFrag s; s.v = wmma_bf16(qf[i], kf, zero8());
#pragma unroll
      for (int r = 0; r < 8; ++r) {
        const int pos = ((r + 8 * lh) << 4) + lc;  // n_local*16 + m_local
        Sh[pos * 34 + h] = f2bf(s.f[r]);
      }
    }
  };

  // ------------------------------ PASS 1 ---------------------------------
  issue_k(0, 0);
  for (int mt = 0; mt < 64; ++mt) {
    if (mt + 1 < 64) {
      issue_k(mt + 1, (mt + 1) & 1);
      ASYNC_WAIT(4);  // oldest 4 (current tile) landed; next 4 in flight
    } else {
      ASYNC_WAIT(0);
    }
    stage_scores(mt & 1);
    __syncthreads();
#pragma unroll
    for (int tI = 0; tI < 2; ++tI) {
      const int t = w * 2 + tI;  // owned n_local
      BFrag sf;                  // B frag: col=m_local=lc, elem e -> h=lh*16+e
      {
        const int base = (t * 16 + lc) * 17 + lh * 8;  // dword index
#pragma unroll
        for (int j = 0; j < 8; ++j) sf.u[j] = Sh32[base + j];
      }
#pragma unroll
      for (int gt = 0; gt < 2; ++gt) {
        FFrag p; p.v = wmma_bf16(pref[gt], sf, zero8());
#pragma unroll
        for (int r = 0; r < 8; ++r) {
          const float v = p.f[r] + bpre_r[gt][r];
          float tm = v;  // row max over the 16 m-lanes of this half
          tm = fmaxf(tm, __shfl_xor(tm, 1, 32));
          tm = fmaxf(tm, __shfl_xor(tm, 2, 32));
          tm = fmaxf(tm, __shfl_xor(tm, 4, 32));
          tm = fmaxf(tm, __shfl_xor(tm, 8, 32));
          const float mOld = runM[tI][gt][r];
          const float mNew = fmaxf(mOld, tm);
          float e = __expf(v - mNew);
          e += __shfl_xor(e, 1, 32);
          e += __shfl_xor(e, 2, 32);
          e += __shfl_xor(e, 4, 32);
          e += __shfl_xor(e, 8, 32);
          runL[tI][gt][r] = runL[tI][gt][r] * __expf(mOld - mNew) + e;
          runM[tI][gt][r] = mNew;
        }
      }
    }
    __syncthreads();
  }

  // finalize: runL becomes 1/L
#pragma unroll
  for (int tI = 0; tI < 2; ++tI)
#pragma unroll
    for (int gt = 0; gt < 2; ++gt)
#pragma unroll
      for (int r = 0; r < 8; ++r) runL[tI][gt][r] = 1.f / runL[tI][gt][r];

  FFrag accv[4];
#pragma unroll
  for (int j = 0; j < 4; ++j)
#pragma unroll
    for (int r = 0; r < 8; ++r) accv[j].f[r] = 0.f;

  // ------------------------------ PASS 2 ---------------------------------
  issue_k(0, 0);
  issue_v(0, 0);
  for (int mt = 0; mt < 64; ++mt) {
    if (mt + 1 < 64) {
      const int nbuf = (mt + 1) & 1;
      issue_k(mt + 1, nbuf);
      issue_v(mt + 1, nbuf);
      ASYNC_WAIT(8);  // oldest 8 (current K+V tiles) landed
    } else {
      ASYNC_WAIT(0);
    }
    const int buf = mt & 1;
    const int m0 = mt * 16;
    (void)m0;
    stage_scores(buf);
    __syncthreads();

    float tmpA[2][2][8];
#pragma unroll
    for (int tI = 0; tI < 2; ++tI) {
      const int t = w * 2 + tI;
      BFrag sf;
      {
        const int base = (t * 16 + lc) * 17 + lh * 8;
#pragma unroll
        for (int j = 0; j < 8; ++j) sf.u[j] = Sh32[base + j];
      }
#pragma unroll
      for (int gt = 0; gt < 2; ++gt) {
        FFrag p; p.v = wmma_bf16(pref[gt], sf, zero8());
#pragma unroll
        for (int r = 0; r < 8; ++r)
          tmpA[tI][gt][r] =
              __expf(p.f[r] + bpre_r[gt][r] - runM[tI][gt][r]) * runL[tI][gt][r];
      }
    }
    __syncthreads();  // all pre-mix reads of Sh done; reuse it for A
#pragma unroll
    for (int tI = 0; tI < 2; ++tI) {
      const int pos = (w * 2 + tI) * 16 + lc;
#pragma unroll
      for (int gt = 0; gt < 2; ++gt)
#pragma unroll
        for (int r = 0; r < 8; ++r)
          Sh[pos * 34 + gt * 16 + 8 * lh + r] = f2bf(tmpA[tI][gt][r]);
    }
    __syncthreads();

    // post-mix:  A~[g'] = Wpost @ A + bpost   -> stage to At[g'][n][m]
#pragma unroll
    for (int tI = 0; tI < 2; ++tI) {
      const int tp = w * 2 + tI;
      BFrag af2;
      {
        const int base = (tp * 16 + lc) * 17 + lh * 8;
#pragma unroll
        for (int j = 0; j < 8; ++j) af2.u[j] = Sh32[base + j];
      }
#pragma unroll
      for (int gt = 0; gt < 2; ++gt) {
        FFrag am; am.v = wmma_bf16(postf[gt], af2, zero8());
#pragma unroll
        for (int r = 0; r < 8; ++r) {
          const int gp = gt * 16 + 8 * lh + r;
          At[(gp * 16 + tp) * 16 + lc] = f2bf(am.f[r] + bpost_r[gt][r]);
        }
      }
    }
    __syncthreads();

    // accumulate:  acc[g'] += A~(n x m) @ V(m x d), V tile from LDS
    const u32* At32 = (const u32*)At;
#pragma unroll
    for (int j = 0; j < 4; ++j) {
      const int gp = w * 4 + j;
      BFrag aA;  // A frag: row = n = lc, elems e<8 -> m = lh*8+e, rest pad 0
      {
        const int base = (gp * 16 + lc) * 8 + lh * 4;  // dword index
#pragma unroll
        for (int q2 = 0; q2 < 4; ++q2) aA.u[q2] = At32[base + q2];
#pragma unroll
        for (int q2 = 4; q2 < 8; ++q2) aA.u[q2] = 0;
      }
      BFrag vf;  // B frag: col = d = lc, elems e<8 -> m = lh*8+e, rest pad 0
      {
        const uint4 t0 = *(const uint4*)&Vt2[buf][gp * 256 + lc * 16 + lh * 8];
        vf.u[0] = t0.x; vf.u[1] = t0.y; vf.u[2] = t0.z; vf.u[3] = t0.w;
        vf.u[4] = 0; vf.u[5] = 0; vf.u[6] = 0; vf.u[7] = 0;
      }
      accv[j].v = wmma_bf16(aA, vf, accv[j].v);
    }
    __syncthreads();
  }

  // write xa[b][n][c = g'*16+d] as bf16 for the projection GEMM
#pragma unroll
  for (int j = 0; j < 4; ++j) {
    const int gp = w * 4 + j;
#pragma unroll
    for (int r = 0; r < 8; ++r) {
      const int n = n0 + r + 8 * lh;
      xab[(b * NN + n) * CC + gp * DD + lc] = f2bf(accv[j].f[r]);
    }
  }
}

// ---------------------------------------------------------------------------
// Kernel 4: output projection.  out = xa @ Wproj^T + bproj  (f32 output)
// ---------------------------------------------------------------------------
__global__ __launch_bounds__(256) void proj_gemm_kernel(
    const u16* __restrict__ xab, const u16* __restrict__ wprojb,
    const float* __restrict__ bproj, float* __restrict__ out) {
  const int w = threadIdx.x >> 5;
  const int lane = threadIdx.x & 31;
  const int lh = lane >> 4, lc = lane & 15;
  const int m0 = blockIdx.x << 4;
  const int nb = w * 64;  // 8 waves * 64 = 512 = full N

  FFrag acc[4];
#pragma unroll
  for (int j = 0; j < 4; ++j)
#pragma unroll
    for (int r = 0; r < 8; ++r) acc[j].f[r] = 0.f;

  for (int k0 = 0; k0 < CC; k0 += 32) {
    BFrag af;
    {
      const uint4 t0 = *(const uint4*)(xab + (m0 + lc) * CC + k0 + lh * 8);
      const uint4 t1 = *(const uint4*)(xab + (m0 + lc) * CC + k0 + 16 + lh * 8);
      af.u[0] = t0.x; af.u[1] = t0.y; af.u[2] = t0.z; af.u[3] = t0.w;
      af.u[4] = t1.x; af.u[5] = t1.y; af.u[6] = t1.z; af.u[7] = t1.w;
    }
#pragma unroll
    for (int j = 0; j < 4; ++j) {
      const int n = nb + j * 16 + lc;
      const u16* pb = wprojb + n * CC + k0 + lh * 8;
      const uint4 b0 = *(const uint4*)pb, b1 = *(const uint4*)(pb + 16);
      BFrag bf;
      bf.u[0] = b0.x; bf.u[1] = b0.y; bf.u[2] = b0.z; bf.u[3] = b0.w;
      bf.u[4] = b1.x; bf.u[5] = b1.y; bf.u[6] = b1.z; bf.u[7] = b1.w;
      acc[j].v = wmma_bf16(af, bf, acc[j].v);
    }
  }

#pragma unroll
  for (int j = 0; j < 4; ++j) {
    const int n = nb + j * 16 + lc;
    const float bias = bproj[n];
#pragma unroll
    for (int r = 0; r < 8; ++r) {
      const int row = m0 + r + 8 * lh;
      out[row * CC + n] = acc[j].f[r] + bias;
    }
  }
}

// ---------------------------------------------------------------------------
// Launch: convert -> qkv gemm -> fused attention -> proj gemm (all on stream)
// Workspace layout (bytes): ~22 MB total
// ---------------------------------------------------------------------------
extern "C" void kernel_launch(void* const* d_in, const int* in_sizes, int n_in,
                              void* d_out, int out_size, void* d_ws,
                              size_t ws_size, hipStream_t stream) {
  const float* x     = (const float*)d_in[0];
  const float* Wqkv  = (const float*)d_in[1];
  const float* bqkv  = (const float*)d_in[2];
  const float* Wpre  = (const float*)d_in[3];
  const float* bpre  = (const float*)d_in[4];
  const float* Wpost = (const float*)d_in[5];
  const float* bpost = (const float*)d_in[6];
  const float* Wproj = (const float*)d_in[7];
  const float* bproj = (const float*)d_in[8];
  float* out = (float*)d_out;

  char* ws = (char*)d_ws;
  u16* xb     = (u16*)(ws + 0);         // 4096*512   bf16 = 4 MB
  u16* wqkvb  = (u16*)(ws + 4194304);   // 1536*512   bf16
  u16* wprojb = (u16*)(ws + 5767168);   // 512*512    bf16
  u16* wpreb  = (u16*)(ws + 6291456);   // 32*32      bf16
  u16* wpostb = (u16*)(ws + 6293504);   // 32*32      bf16
  u16* qb     = (u16*)(ws + 6295552);   // [b][h][n][d] bf16, pre-scaled
  u16* kb     = (u16*)(ws + 10489856);  // [b][h][n][d] bf16
  u16* vt     = (u16*)(ws + 14684160);  // [b][h][d][n] bf16 (transposed)
  u16* xab    = (u16*)(ws + 18878464);  // [b][n][c]  bf16

  convert_kernel<<<2048, 256, 0, stream>>>(x, Wqkv, Wproj, Wpre, Wpost, xb,
                                           wqkvb, wprojb, wpreb, wpostb);
  qkv_gemm_kernel<<<dim3(256, 3), 256, 0, stream>>>(xb, wqkvb, bqkv, qb, kb, vt);
  attn_kernel<<<256, 256, 0, stream>>>(qb, kb, vt, wpreb, wpostb, bpre, bpost,
                                       xab);
  proj_gemm_kernel<<<dim3(256, 1), 256, 0, stream>>>(xab, wprojb, bproj, out);
}
